// FlowDecoder_18872086299255
// MI455X (gfx1250) — matrix operand order repaired and checked by reading, not compiled, not run on hardware
//
#include <hip/hip_runtime.h>
#include <math.h>

// ---------------------------------------------------------------------------
// MI455X (gfx1250) implementation of the 3-level RAFT-style flow decoder.
//
// Conv/GEMM work runs through one implicit-GEMM kernel built on
// v_wmma_f32_16x16x32_f16 (wave32, 16x16 C/D tile, f32 accumulate):
//   * 16(M) x 64(N) register blocking per wave -> 4 WMMAs per K-chunk,
//   * weight tile double-buffered in LDS by the Tensor Data Mover
//     (tensor_load_to_lds + s_wait_tensorcnt ping-pong),
//   * incremental (ci,kh,kw) im2col decode shared across the 4 B-fragments.
//
// The correlation pyramid is NEVER materialized: avg-pooling the corr volume
// over the f2 spatial dims is linear in f2, so we pool the projected f2
// features and evaluate the 25-point lookup as on-the-fly channel dots.
//
// Input order (setup_inputs dict order, params flattened as a JAX pytree,
// i.e. dict keys sorted at every level):
//   0 feat_render_0  1 feat_real_0  2 feat_render_1  3 feat_real_1
//   4 feat_render_2  5 feat_real_2  6 init_flow      7 init_certainty
//   8+28*lvl + :
//     enc:   0 corr_b1 1 corr_b2 2 corr_w1 3 corr_w2 4 flow_b1 5 flow_b2
//            6 flow_w1 7 flow_w2 8 out_b   9 out_w
//     flowh: 10 b1 11 b2 12 bp 13 w1 14 w2 15 wp
//     maskh: 16 b1 17 b2 18 bp 19 w1 20 w2 21 wp
//     proj:  22 b 23 bn_b 24 bn_g 25 bn_m 26 bn_v 27 w
//
// Workspace requirement: ~155 MB (see layout in kernel_launch).
// ---------------------------------------------------------------------------

typedef __attribute__((ext_vector_type(16))) _Float16 v16h;
typedef __attribute__((ext_vector_type(8)))  float    v8f;
typedef __attribute__((ext_vector_type(4)))  unsigned v4u;
typedef __attribute__((ext_vector_type(8)))  int      v8i;
typedef __attribute__((ext_vector_type(4)))  int      v4i;

#if __has_builtin(__builtin_amdgcn_tensor_load_to_lds)
#define HAVE_TDM 1
#else
#define HAVE_TDM 0
#endif

#if HAVE_TDM
// Issue a 2D TDM load of a rows x cols f32 tile (row stride = rowstride
// elements) into LDS at byte offset lds_off. tensor_dim0/1 are the remaining
// extents from the tile origin so the TDM zero-fills out-of-range reads
// (tail K-chunks / tail Cout tiles).
__device__ __forceinline__ void tdm_load_tile_f32(unsigned lds_off, const float* gptr,
                                                  int tdim0, int tdim1, int rowstride,
                                                  int cols, int rows)
{
    unsigned long long ga = (unsigned long long)(size_t)gptr;
    v4u g0;
    g0[0] = 1u;                                            // count=1, user desc
    g0[1] = lds_off;                                       // lds_addr (bytes)
    g0[2] = (unsigned)(ga & 0xffffffffull);                // global_addr lo
    g0[3] = (unsigned)((ga >> 32) & 0x01ffffffull)         // global_addr hi (57b)
          | (2u << 30);                                    // type=2 ("image")
    v8i g1;
    g1[0] = 0x00020000;                                    // data_size=2 (4B), no mcast
    g1[1] = (int)(((unsigned)tdim0 & 0xffffu) << 16);      // tensor_dim0[15:0]
    g1[2] = (int)((((unsigned)tdim0 >> 16) & 0xffffu)      // tensor_dim0[31:16]
          | (((unsigned)tdim1 & 0xffffu) << 16));          // tensor_dim1[15:0]
    g1[3] = (int)((((unsigned)tdim1 >> 16) & 0xffffu)      // tensor_dim1[31:16]
          | (((unsigned)cols & 0xffffu) << 16));           // tile_dim0
    g1[4] = (int)((unsigned)rows & 0xffffu);               // tile_dim1 (tile_dim2=0)
    g1[5] = rowstride;                                     // tensor_dim0_stride lo32
    g1[6] = 0;                                             // stride hi / dim1_stride lo
    g1[7] = 0;
    v4i gz = {0, 0, 0, 0};
#if defined(__clang_major__) && (__clang_major__ >= 23)
    v8i gz8 = {0, 0, 0, 0, 0, 0, 0, 0};
    __builtin_amdgcn_tensor_load_to_lds(g0, g1, gz, gz, gz8, 0);
#else
    __builtin_amdgcn_tensor_load_to_lds(g0, g1, gz, gz, 0);
#endif
}
#endif

__device__ __forceinline__ void wait_dscnt0()
{
#if __has_builtin(__builtin_amdgcn_s_wait_dscnt)
    __builtin_amdgcn_s_wait_dscnt(0);
#else
    asm volatile("s_wait_dscnt 0" ::: "memory");
#endif
}

// -------------------------- WMMA implicit-GEMM conv ------------------------
// One wave computes a 16(Cout) x 64(HW) output tile; K = Cin*KH*KW in chunks
// of 32 fed to v_wmma_f32_16x16x32_f16 (four accumulators share the A tile).
__global__ __launch_bounds__(32)
void k_conv_wmma(const float* __restrict__ X, const float* __restrict__ Wt,
                 const float* __restrict__ scale, const float* __restrict__ shift,
                 float* __restrict__ Y,
                 int Cin, int H, int Wd, int Cout, int KH, int KW, int pad,
                 int relu, int out_cstride, int out_coff)
{
    __shared__ float lds_w[2][16 * 32];         // double-buffered 16x32 f32 weight tile

    const int lane = threadIdx.x;
    const int HWn  = H * Wd;
    const int n0   = blockIdx.x * 64;           // spatial tile (four 16-wide groups)
    const int m0   = blockIdx.y * 16;           // Cout tile
    const int b    = blockIdx.z;
    const int KK   = KH * KW;
    const int K    = Cin * KK;
    const int nchunks = (K + 31) / 32;

    const int nIdx0 = n0 + (lane & 15);
    const int nIdx1 = nIdx0 + 16;
    const int nIdx2 = nIdx0 + 32;
    const int nIdx3 = nIdx0 + 48;
    const int oh0 = nIdx0 / Wd, ow0 = nIdx0 - oh0 * Wd;
    const int oh1 = nIdx1 / Wd, ow1 = nIdx1 - oh1 * Wd;
    const int oh2 = nIdx2 / Wd, ow2 = nIdx2 - oh2 * Wd;
    const int oh3 = nIdx3 / Wd, ow3 = nIdx3 - oh3 * Wd;

    // A lane mapping: lanes 0-15 -> M=lane, K elems {0..15};
    //                 lanes 16-31 -> M=lane-16, K elems {16..31}
    const int  mA     = m0 + (lane & 15);
    const bool mvalid = (mA < Cout);
    const float* wrow = Wt + (size_t)mA * (size_t)K;
    const int kbaseA  = (lane >> 4) * 16;
    const int rowA    = lane & 15;
    // B lane mapping: lanes 0-15 -> K 0..15, lanes 16-31 -> K 16..31
    const int kbaseB  = (lane >> 4) * 16;
    (void)wrow; (void)mvalid; (void)rowA;

#if HAVE_TDM
    const unsigned lds_off0 = (unsigned)(size_t)(&lds_w[0][0]);
    const unsigned lds_off1 = (unsigned)(size_t)(&lds_w[1][0]);
    // Prologue: stage chunk 0 into buffer 0.
    tdm_load_tile_f32(lds_off0, Wt + (size_t)m0 * K,
                      /*tdim0=*/K, /*tdim1=*/Cout - m0,
                      /*rowstride=*/K, /*cols=*/32, /*rows=*/16);
#endif

    v8f acc0 = {}, acc1 = {}, acc2 = {}, acc3 = {};

    for (int c = 0; c < nchunks; ++c) {
        const int k0 = c * 32;
        __builtin_prefetch(Wt + (size_t)m0 * K + k0 + 64, 0, 0);  // global_prefetch_b8

        v16h a;
#if HAVE_TDM
        const bool hasNext = (c + 1 < nchunks);
        if (hasNext) {
            wait_dscnt0();                       // prior LDS reads of the other buffer done
            tdm_load_tile_f32((c & 1) ? lds_off0 : lds_off1,
                              Wt + (size_t)m0 * K + (k0 + 32),
                              /*tdim0=*/K - (k0 + 32), /*tdim1=*/Cout - m0,
                              /*rowstride=*/K, /*cols=*/32, /*rows=*/16);
            __builtin_amdgcn_s_wait_tensorcnt((short)1);   // current tile resident
        } else {
            __builtin_amdgcn_s_wait_tensorcnt((short)0);
        }
        const float* lw = &lds_w[c & 1][0];
#pragma unroll
        for (int e = 0; e < 16; ++e) {
            int kc = kbaseA + e;                           // 0..31 within tile
            a[e] = (_Float16)lw[rowA * 32 + kc];           // ds_load; TDM zero-fills OOB
        }
#else
#pragma unroll
        for (int e = 0; e < 16; ++e) {
            int k   = k0 + kbaseA + e;
            float v = (mvalid && k < K) ? wrow[k] : 0.0f;
            a[e] = (_Float16)v;
        }
#endif

        // im2col B fragments: shared incremental (ci,kh,kw) decode, 4 pixel groups.
        v16h bm0, bm1, bm2, bm3;
        {
            int kk = k0 + kbaseB;
            int ci = kk / KK;
            int r  = kk - ci * KK;
            int kh = r / KW;
            int kw = r - kh * KW;
#pragma unroll
            for (int e = 0; e < 16; ++e) {
                float v0 = 0.f, v1 = 0.f, v2 = 0.f, v3 = 0.f;
                if (ci < Cin) {                              // k < K
                    const float* xc = X + ((size_t)b * Cin + ci) * (size_t)HWn;
                    int dh = kh - pad, dw = kw - pad;
                    int ih, iw;
                    ih = oh0 + dh; iw = ow0 + dw;
                    if ((unsigned)ih < (unsigned)H && (unsigned)iw < (unsigned)Wd)
                        v0 = xc[ih * Wd + iw];
                    ih = oh1 + dh; iw = ow1 + dw;
                    if ((unsigned)ih < (unsigned)H && (unsigned)iw < (unsigned)Wd)
                        v1 = xc[ih * Wd + iw];
                    ih = oh2 + dh; iw = ow2 + dw;
                    if ((unsigned)ih < (unsigned)H && (unsigned)iw < (unsigned)Wd)
                        v2 = xc[ih * Wd + iw];
                    ih = oh3 + dh; iw = ow3 + dw;
                    if ((unsigned)ih < (unsigned)H && (unsigned)iw < (unsigned)Wd)
                        v3 = xc[ih * Wd + iw];
                }
                bm0[e] = (_Float16)v0;
                bm1[e] = (_Float16)v1;
                bm2[e] = (_Float16)v2;
                bm3[e] = (_Float16)v3;
                // advance (ci,kh,kw) by one k
                if (++kw == KW) { kw = 0; if (++kh == KH) { kh = 0; ++ci; } }
            }
        }

        acc0 = __builtin_amdgcn_wmma_f32_16x16x32_f16(false, a, false, bm0,
                                                      (short)0, acc0, false, false);
        acc1 = __builtin_amdgcn_wmma_f32_16x16x32_f16(false, a, false, bm1,
                                                      (short)0, acc1, false, false);
        acc2 = __builtin_amdgcn_wmma_f32_16x16x32_f16(false, a, false, bm2,
                                                      (short)0, acc2, false, false);
        acc3 = __builtin_amdgcn_wmma_f32_16x16x32_f16(false, a, false, bm3,
                                                      (short)0, acc3, false, false);
    }

    // C/D layout: VGPR r: lanes 0-15 -> M=r, lanes 16-31 -> M=r+8; N = lane&15
    const int mw = m0 + ((lane >> 4) << 3);
#pragma unroll
    for (int r = 0; r < 8; ++r) {
        int m = mw + r;
        if (m < Cout) {
            float sc = scale ? scale[m] : 1.0f;
            float sh = shift ? shift[m] : 0.0f;
            float v0 = acc0[r] * sc + sh;
            float v1 = acc1[r] * sc + sh;
            float v2 = acc2[r] * sc + sh;
            float v3 = acc3[r] * sc + sh;
            if (relu) {
                v0 = v0 > 0.0f ? v0 : 0.0f;
                v1 = v1 > 0.0f ? v1 : 0.0f;
                v2 = v2 > 0.0f ? v2 : 0.0f;
                v3 = v3 > 0.0f ? v3 : 0.0f;
            }
            float* yr = Y + ((size_t)b * out_cstride + out_coff + m) * (size_t)HWn;
            yr[nIdx0] = v0;
            yr[nIdx1] = v1;
            yr[nIdx2] = v2;
            yr[nIdx3] = v3;
        }
    }
}

// --------------------------- BN fold for proj ------------------------------
__global__ void k_bn_prep(const float* __restrict__ bias, const float* __restrict__ g,
                          const float* __restrict__ bb,   const float* __restrict__ m,
                          const float* __restrict__ v,
                          float* __restrict__ scale, float* __restrict__ shift, int C)
{
    int c = blockIdx.x * blockDim.x + threadIdx.x;
    if (c >= C) return;
    float s = g[c] * rsqrtf(v[c] + 1e-5f);
    scale[c] = s;
    shift[c] = (bias[c] - m[c]) * s + bb[c];
}

// ------------------------------- 2x2 avg pool ------------------------------
__global__ void k_avgpool2(const float* __restrict__ src, float* __restrict__ dst,
                           int Bn, int C, int H, int Wd)
{
    int Ho = H >> 1, Wo = Wd >> 1;
    long total = (long)Bn * C * Ho * Wo;
    long idx = (long)blockIdx.x * blockDim.x + threadIdx.x;
    if (idx >= total) return;
    int x = idx % Wo; long t = idx / Wo;
    int y = t % Ho;  t /= Ho;
    int c = t % C;   int b = t / C;
    const float* s = src + (((size_t)b * C + c) * H) * Wd;
    float v = 0.25f * (s[(2*y)*Wd + 2*x] + s[(2*y)*Wd + 2*x + 1] +
                       s[(2*y+1)*Wd + 2*x] + s[(2*y+1)*Wd + 2*x + 1]);
    dst[(((size_t)b * C + c) * Ho + y) * Wo + x] = v;
}

// -------------------- correlation lookup (pooled-f2 trick) -----------------
// out[b, lv*25+kk, p] = (1/16) * sum_c fr[b,c,p] * bilin(fe_pool_lv[b,c], sx, sy)
// with sx = (gx+flow_x)/2^lv + (kk/5 - 2), sy = (gy+flow_y)/2^lv + (kk%5 - 2)
// (mmflow quirk: the dy-meshgrid channel is added to x).
__global__ void k_corr_lookup(const float* __restrict__ fr,
                              const float* __restrict__ f0,
                              const float* __restrict__ f1p,
                              const float* __restrict__ f2p,
                              const float* __restrict__ flow,
                              float* __restrict__ out,
                              int Bn, int C, int H, int Wd, int nl)
{
    const int HWn = H * Wd;
    long total = (long)Bn * HWn * nl * 25;
    long idx = (long)blockIdx.x * blockDim.x + threadIdx.x;
    if (idx >= total) return;
    int kk = idx % 25; long t = idx / 25;
    int lv = t % nl;   t /= nl;
    int p  = t % HWn;  int b = t / HWn;

    float gx = (float)(p % Wd);
    float gy = (float)(p / Wd);
    float cx = gx + flow[((size_t)b * 2 + 0) * HWn + p];
    float cy = gy + flow[((size_t)b * 2 + 1) * HWn + p];
    float inv = 1.0f / (float)(1 << lv);
    float sx = cx * inv + (float)(kk / 5) - 2.0f;
    float sy = cy * inv + (float)(kk % 5) - 2.0f;

    int Hl = H >> lv, Wl = Wd >> lv;
    const float* fe = (lv == 0) ? f0 : ((lv == 1) ? f1p : f2p);

    int x0 = (int)floorf(sx), y0 = (int)floorf(sy);
    float fx = sx - (float)x0, fy = sy - (float)y0;
    float w00 = (1.f - fx) * (1.f - fy), w10 = fx * (1.f - fy);
    float w01 = (1.f - fx) * fy,         w11 = fx * fy;
    bool vx0 = (x0 >= 0) && (x0 <= Wl - 1);
    bool vx1 = (x0 + 1 >= 0) && (x0 + 1 <= Wl - 1);
    bool vy0 = (y0 >= 0) && (y0 <= Hl - 1);
    bool vy1 = (y0 + 1 >= 0) && (y0 + 1 <= Hl - 1);
    w00 *= (vx0 && vy0); w10 *= (vx1 && vy0);
    w01 *= (vx0 && vy1); w11 *= (vx1 && vy1);
    int xc0 = min(max(x0, 0), Wl - 1),     yc0 = min(max(y0, 0), Hl - 1);
    int xc1 = min(max(x0 + 1, 0), Wl - 1), yc1 = min(max(y0 + 1, 0), Hl - 1);

    const float* frb = fr + (size_t)b * C * HWn + p;
    const float* feb = fe + (size_t)b * C * Hl * Wl;
    float acc = 0.0f;
    for (int c = 0; c < C; ++c) {
        const float* fc = feb + (size_t)c * Hl * Wl;
        float s = w00 * fc[yc0 * Wl + xc0] + w10 * fc[yc0 * Wl + xc1]
                + w01 * fc[yc1 * Wl + xc0] + w11 * fc[yc1 * Wl + xc1];
        acc += frb[(size_t)c * HWn] * s;
    }
    out[((size_t)b * (25 * nl) + lv * 25 + kk) * (size_t)HWn + p] = acc * 0.0625f;
}

// ----------------------- bilinear feature warp (fe_hat) --------------------
__global__ void k_feature_sample(const float* __restrict__ fe, const float* __restrict__ flow,
                                 float* __restrict__ dst,
                                 int Bn, int C, int H, int Wd, int out_cstride, int out_coff)
{
    const int HWn = H * Wd;
    long total = (long)Bn * C * HWn;
    long idx = (long)blockIdx.x * blockDim.x + threadIdx.x;
    if (idx >= total) return;
    int p = idx % HWn; long t = idx / HWn;
    int c = t % C;     int b = t / C;

    float sx = (float)(p % Wd) + flow[((size_t)b * 2 + 0) * HWn + p];
    float sy = (float)(p / Wd) + flow[((size_t)b * 2 + 1) * HWn + p];
    int x0 = (int)floorf(sx), y0 = (int)floorf(sy);
    float fx = sx - (float)x0, fy = sy - (float)y0;
    float w00 = (1.f - fx) * (1.f - fy), w10 = fx * (1.f - fy);
    float w01 = (1.f - fx) * fy,         w11 = fx * fy;
    w00 *= (x0 >= 0 && x0 < Wd && y0 >= 0 && y0 < H);
    w10 *= (x0 + 1 >= 0 && x0 + 1 < Wd && y0 >= 0 && y0 < H);
    w01 *= (x0 >= 0 && x0 < Wd && y0 + 1 >= 0 && y0 + 1 < H);
    w11 *= (x0 + 1 >= 0 && x0 + 1 < Wd && y0 + 1 >= 0 && y0 + 1 < H);
    int xc0 = min(max(x0, 0), Wd - 1),     yc0 = min(max(y0, 0), H - 1);
    int xc1 = min(max(x0 + 1, 0), Wd - 1), yc1 = min(max(y0 + 1, 0), H - 1);

    const float* fc = fe + ((size_t)b * C + c) * HWn;
    float v = w00 * fc[yc0 * Wd + xc0] + w10 * fc[yc0 * Wd + xc1]
            + w01 * fc[yc1 * Wd + xc0] + w11 * fc[yc1 * Wd + xc1];
    dst[((size_t)b * out_cstride + out_coff + c) * (size_t)HWn + p] = v;
}

// ---------------------- channel copy into strided buffer -------------------
__global__ void k_copy_ch(const float* __restrict__ src, float* __restrict__ dst,
                          int Bn, int C, int HWn, int dst_cstride, int dst_coff)
{
    long total = (long)Bn * C * HWn;
    long idx = (long)blockIdx.x * blockDim.x + threadIdx.x;
    if (idx >= total) return;
    int p = idx % HWn; long t = idx / HWn;
    int c = t % C;     int b = t / C;
    dst[((size_t)b * dst_cstride + dst_coff + c) * (size_t)HWn + p] =
        src[((size_t)b * C + c) * (size_t)HWn + p];
}

// ------------------------------ y += d -------------------------------------
__global__ void k_add(float* __restrict__ y, const float* __restrict__ d, long n)
{
    long idx = (long)blockIdx.x * blockDim.x + threadIdx.x;
    if (idx < n) y[idx] += d[idx];
}

// --------------- bilinear 2x upsample (align_corners=True) -----------------
__global__ void k_upsample2(const float* __restrict__ src, float* __restrict__ dst,
                            int Bn, int C, int H, int Wd, float vscale)
{
    int Ho = 2 * H, Wo = 2 * Wd;
    long total = (long)Bn * C * Ho * Wo;
    long idx = (long)blockIdx.x * blockDim.x + threadIdx.x;
    if (idx >= total) return;
    int x = idx % Wo; long t = idx / Wo;
    int y = t % Ho;  t /= Ho;
    int c = t % C;   int b = t / C;

    float px = (float)x * (float)(Wd - 1) / (float)(Wo - 1);
    float py = (float)y * (float)(H - 1) / (float)(Ho - 1);
    int x0 = (int)floorf(px), y0 = (int)floorf(py);
    int x1 = min(x0 + 1, Wd - 1), y1 = min(y0 + 1, H - 1);
    float fx = px - (float)x0, fy = py - (float)y0;

    const float* s = src + ((size_t)b * C + c) * (size_t)(H * Wd);
    float v = (1.f - fy) * ((1.f - fx) * s[y0 * Wd + x0] + fx * s[y0 * Wd + x1])
            + fy         * ((1.f - fx) * s[y1 * Wd + x0] + fx * s[y1 * Wd + x1]);
    dst[(((size_t)b * C + c) * Ho + y) * Wo + x] = v * vscale;
}

// ---------------------------------------------------------------------------
extern "C" void kernel_launch(void* const* d_in, const int* in_sizes, int n_in,
                              void* d_out, int out_size, void* d_ws, size_t ws_size,
                              hipStream_t stream)
{
    (void)in_sizes; (void)n_in; (void)out_size; (void)ws_size;
    const int Bn = 4;

    const float* fR[3] = { (const float*)d_in[0], (const float*)d_in[2], (const float*)d_in[4] };
    const float* fE[3] = { (const float*)d_in[1], (const float*)d_in[3], (const float*)d_in[5] };
    const float* init_flow = (const float*)d_in[6];
    const float* init_cert = (const float*)d_in[7];
    auto P = [&](int lvl, int off) -> const float* {
        return (const float*)d_in[8 + 28 * lvl + off];
    };
    float* out = (float*)d_out;

    // ------------------------- workspace layout ---------------------------
    const size_t MHW = 4096;                  // 64*64 (level 2)
    float* w = (float*)d_ws;
    size_t o = 0;
    auto alloc = [&](size_t n) { float* p = w + o; o += n; return p; };
    float* fr    = alloc((size_t)Bn * 256 * MHW);        // projected renders
    float* fe    = alloc((size_t)Bn * 256 * MHW);        // projected reals
    float* fp1   = alloc((size_t)Bn * 256 * (MHW / 4));  // fe pooled 1x
    float* fp2   = alloc((size_t)Bn * 256 * (MHW / 16)); // fe pooled 2x
    float* corr  = alloc((size_t)Bn * 75  * MHW);        // lookup output
    float* catcf = alloc((size_t)Bn * 256 * MHW);        // concat[c2(192), f2(64)]
    float* xbuf  = alloc((size_t)Bn * 640 * MHW);        // concat[fr, fe_hat, motion]
    float* h1    = alloc((size_t)Bn * 512 * MHW);        // head hidden 1 / enc c1
    float* h2    = alloc((size_t)Bn * 256 * MHW);        // head hidden 2 / enc f1
    float* flowA = alloc((size_t)Bn * 2 * MHW);
    float* flowB = alloc((size_t)Bn * 2 * MHW);
    float* certA = alloc((size_t)Bn * 1 * MHW);
    float* certB = alloc((size_t)Bn * 1 * MHW);
    float* dflow = alloc((size_t)Bn * 2 * MHW);
    float* dcert = alloc((size_t)Bn * 1 * MHW);
    float* bnsc  = alloc(256);
    float* bnsh  = alloc(256);

    auto conv = [&](const float* X, const float* Wt, const float* sc, const float* sh,
                    float* Y, int Cin, int H, int Wd, int Cout, int KH, int KW, int pad,
                    int relu, int ostride, int ooff) {
        dim3 g((unsigned)((H * Wd) / 64), (unsigned)((Cout + 15) / 16), (unsigned)Bn);
        k_conv_wmma<<<g, 32, 0, stream>>>(X, Wt, sc, sh, Y, Cin, H, Wd, Cout,
                                          KH, KW, pad, relu, ostride, ooff);
    };
    auto gl = [](long n) { return (unsigned)((n + 255) / 256); };

    float* flow = flowA; float* flowN = flowB;
    float* cert = certA; float* certN = certB;

    // init flow / certainty (level-0 resolution 16x16)
    k_copy_ch<<<gl((long)Bn * 2 * 256), 256, 0, stream>>>(init_flow, flow, Bn, 2, 256, 2, 0);
    k_copy_ch<<<gl((long)Bn * 1 * 256), 256, 0, stream>>>(init_cert, cert, Bn, 1, 256, 1, 0);

    const long pf_off[3] = { 0, 2048, 10240 };
    const long pc_off[3] = { 43008, 44032, 48128 };

    for (int lvl = 0; lvl < 3; ++lvl) {
        const int H = 16 << lvl, Wd = 16 << lvl;
        const int HW = H * Wd;
        const int nl = lvl + 1;            // corr pyramid depth
        const int cc = 25 * nl;            // correlation channels

        // ---- projections (1x1 conv + folded BN) ----
        k_bn_prep<<<1, 256, 0, stream>>>(P(lvl,22), P(lvl,24), P(lvl,23),
                                         P(lvl,25), P(lvl,26), bnsc, bnsh, 256);
        conv(fR[lvl], P(lvl,27), bnsc, bnsh, fr, 256, H, Wd, 256, 1, 1, 0, 0, 256, 0);
        conv(fE[lvl], P(lvl,27), bnsc, bnsh, fe, 256, H, Wd, 256, 1, 1, 0, 0, 256, 0);

        // ---- f2 feature pyramid (replaces corr-volume pooling) ----
        if (nl >= 2) k_avgpool2<<<gl((long)Bn*256*(HW/4)), 256, 0, stream>>>(fe,  fp1, Bn, 256, H,   Wd);
        if (nl >= 3) k_avgpool2<<<gl((long)Bn*256*(HW/16)),256, 0, stream>>>(fp1, fp2, Bn, 256, H/2, Wd/2);

        // ---- correlation lookup ----
        k_corr_lookup<<<gl((long)Bn * HW * cc), 256, 0, stream>>>(
            fr, fe, fp1, fp2, flow, corr, Bn, 256, H, Wd, nl);

        // ---- motion encoder ----
        conv(corr, P(lvl,2), nullptr, P(lvl,0), h1,    cc,  H, Wd, 256, 1, 1, 0, 1, 256, 0);
        conv(h1,   P(lvl,3), nullptr, P(lvl,1), catcf, 256, H, Wd, 192, 3, 3, 1, 1, 256, 0);
        conv(flow, P(lvl,6), nullptr, P(lvl,4), h2,    2,   H, Wd, 128, 7, 7, 3, 1, 128, 0);
        conv(h2,   P(lvl,7), nullptr, P(lvl,5), catcf, 128, H, Wd, 64,  3, 3, 1, 1, 256, 192);
        conv(catcf,P(lvl,9), nullptr, P(lvl,8), xbuf,  256, H, Wd, 126, 3, 3, 1, 1, 640, 512);
        k_copy_ch<<<gl((long)Bn * 2 * HW), 256, 0, stream>>>(flow, xbuf, Bn, 2, HW, 640, 638);

        // ---- big concat: [fr | fe_hat | motion] ----
        k_copy_ch<<<gl((long)Bn * 256 * HW), 256, 0, stream>>>(fr, xbuf, Bn, 256, HW, 640, 0);
        k_feature_sample<<<gl((long)Bn * 256 * HW), 256, 0, stream>>>(
            fe, flow, xbuf, Bn, 256, H, Wd, 640, 256);

        // ---- flow head ----
        conv(xbuf, P(lvl,13), nullptr, P(lvl,10), h1,    640, H, Wd, 512, 3, 3, 1, 1, 512, 0);
        conv(h1,   P(lvl,14), nullptr, P(lvl,11), h2,    512, H, Wd, 256, 3, 3, 1, 1, 256, 0);
        conv(h2,   P(lvl,15), nullptr, P(lvl,12), dflow, 256, H, Wd, 2,   3, 3, 1, 0, 2,   0);
        // ---- mask head ----
        conv(xbuf, P(lvl,19), nullptr, P(lvl,16), h1,    640, H, Wd, 512, 3, 3, 1, 1, 512, 0);
        conv(h1,   P(lvl,20), nullptr, P(lvl,17), h2,    512, H, Wd, 256, 3, 3, 1, 1, 256, 0);
        conv(h2,   P(lvl,21), nullptr, P(lvl,18), dcert, 256, H, Wd, 1,   1, 1, 0, 0, 1,   0);

        k_add<<<gl((long)Bn * 2 * HW), 256, 0, stream>>>(flow, dflow, (long)Bn * 2 * HW);
        k_add<<<gl((long)Bn * 1 * HW), 256, 0, stream>>>(cert, dcert, (long)Bn * 1 * HW);

        // ---- write outputs ----
        k_copy_ch<<<gl((long)Bn * 2 * HW), 256, 0, stream>>>(flow, out + pf_off[lvl], Bn, 2, HW, 2, 0);
        k_copy_ch<<<gl((long)Bn * 1 * HW), 256, 0, stream>>>(cert, out + pc_off[lvl], Bn, 1, HW, 1, 0);

        // ---- upsample to next level ----
        if (lvl != 2) {
            k_upsample2<<<gl((long)Bn * 2 * 4 * HW), 256, 0, stream>>>(flow, flowN, Bn, 2, H, Wd, 2.0f);
            k_upsample2<<<gl((long)Bn * 1 * 4 * HW), 256, 0, stream>>>(cert, certN, Bn, 1, H, Wd, 1.0f);
            float* t;
            t = flow; flow = flowN; flowN = t;
            t = cert; cert = certN; certN = t;
        }
    }
}
